// HungarianMatcher_8615704396256
// MI455X (gfx1250) — matrix operand order repaired and checked by reading, not compiled, not run on hardware
//
#include <hip/hip_runtime.h>

typedef __attribute__((ext_vector_type(16))) _Float16 v16h;
typedef __attribute__((ext_vector_type(8)))  float    v8f;

namespace {
constexpr int B = 2, NQ = 400, NCLS = 19, NINST = 32, NPTS = 100000;
constexpr int MTILES = NQ / 16;     // 25
constexpr int KSPLIT = 63;          // 62 full chunks of 1600 + tail of 800 (both /32)
constexpr int KCHUNK = 1600;        // multiple of 32
constexpr int PFDIST = 512;         // prefetch distance in floats (16 steps ahead)
// fp32 workspace layout
constexpr int ACC_OFF = 0;                      // [B][NQ][NINST][3] : pm.tm, sig.tm, bin.tm
constexpr int ACC_N   = B * NQ * NINST * 3;
constexpr int ROW_OFF = ACC_OFF + ACC_N;        // [B][NQ][3] : sum(neg), sum(sig), sum(bin)
constexpr int ROW_N   = B * NQ * 3;
constexpr int TMR_OFF = ROW_OFF + ROW_N;        // [B][NINST] : sum(tm)
constexpr int TMR_N   = B * NINST;
constexpr int WS_FLOATS = TMR_OFF + TMR_N;
// optional f16 copy of tm appended after the fp32 region (16B aligned)
constexpr long TMH_OFF_F   = (WS_FLOATS + 3) & ~3L;            // float index, 16B aligned
constexpr long TMH_ELEMS   = (long)B * NINST * NPTS;           // halfs
constexpr long WS_REQ_TMH  = TMH_OFF_F * 4 + TMH_ELEMS * 2;    // bytes needed for f16 path
}

__global__ void hm_ws_zero(float* __restrict__ ws) {
  int i = blockIdx.x * blockDim.x + threadIdx.x;
  if (i < WS_FLOATS) ws[i] = 0.0f;
}

// One block per (b, inst): row-sum of tm, and (optionally) convert the row to f16.
__global__ __launch_bounds__(256) void hm_tm_prep(const float* __restrict__ tm,
                                                  float* __restrict__ ws,
                                                  _Float16* __restrict__ tmh) {
  __shared__ float red[256];
  const int bm = blockIdx.x;                     // b*NINST + inst
  const float* rowp = tm + (long)bm * NPTS;
  _Float16* hrow = tmh ? (tmh + (long)bm * NPTS) : nullptr;
  float s = 0.0f;
  for (int i = threadIdx.x; i < NPTS; i += 256) {
    float v = rowp[i];
    s += v;
    if (hrow) hrow[i] = (_Float16)v;
  }
  red[threadIdx.x] = s;
  __syncthreads();
  for (int off = 128; off > 0; off >>= 1) {
    if (threadIdx.x < off) red[threadIdx.x] += red[threadIdx.x + off];
    __syncthreads();
  }
  if (threadIdx.x == 0) ws[TMR_OFF + bm] = red[0];
}

// One wave per block: 16-query M-tile x full NINST, over one K-chunk.
// USE_TMH: load B-tiles directly as f16 (preconverted) vs fp32 + inline convert.
template <bool USE_TMH>
__global__ __launch_bounds__(32) void hm_wmma_kernel(const float* __restrict__ pm,
                                                     const float* __restrict__ tm,
                                                     const _Float16* __restrict__ tmh,
                                                     float* __restrict__ ws) {
  const int lane = threadIdx.x;
  const int r    = lane & 15;
  const int half = lane >> 4;

  int bx = blockIdx.x;
  const int ks = bx % KSPLIT; bx /= KSPLIT;
  const int mt = bx % MTILES; bx /= MTILES;
  const int b  = bx;

  const int q = mt * 16 + r;                     // this lane's A-matrix row (query)
  long kbeg = (long)ks * KCHUNK;
  long kend = kbeg + KCHUNK; if (kend > NPTS) kend = NPTS;

  const float*    pmRow   = pm  + (long)(b * NQ + q) * NPTS;
  const float*    tmRow0  = tm  + (long)(b * NINST + r) * NPTS;        // N-tile 0: inst r
  const float*    tmRow1  = tm  + (long)(b * NINST + 16 + r) * NPTS;   // N-tile 1: inst 16+r
  const _Float16* tmhRow0 = USE_TMH ? tmh + (long)(b * NINST + r) * NPTS : nullptr;
  const _Float16* tmhRow1 = USE_TMH ? tmh + (long)(b * NINST + 16 + r) * NPTS : nullptr;

  // A k order used here (contiguous, same as B): lanes 0-15 hold K 0..15; lanes 16-31 hold K 16..31
  const int aoff = half ? 16 : 0;
  // B k order used here (contiguous, same as A): lanes 0-15 hold K 0..15; lanes 16-31 hold K 16..31
  const int boff = half ? 16 : 0;

  v8f acc_pm0{}, acc_pm1{}, acc_sg0{}, acc_sg1{}, acc_bn0{}, acc_bn1{};
  float neg_s = 0.0f, sig_s = 0.0f, bin_s = 0.0f;

  for (long k = kbeg; k < kend; k += 32) {
    // keep the HBM stream ahead of the consume point (global_prefetch_b8)
    if (k + PFDIST < NPTS) __builtin_prefetch(pmRow + k + PFDIST, 0, 0);

    // ---- load 16 pm values for this lane (one contiguous run of 16 floats)
    const float4* pa = reinterpret_cast<const float4*>(pmRow + k + aoff);
    float4 a0 = pa[0], a1 = pa[1], a2 = pa[2], a3 = pa[3];
    float xv[16] = { a0.x,a0.y,a0.z,a0.w, a1.x,a1.y,a1.z,a1.w,
                     a2.x,a2.y,a2.z,a2.w, a3.x,a3.y,a3.z,a3.w };

    v16h Apm, Asg, Abn;
#pragma unroll
    for (int i = 0; i < 16; ++i) {
      float x  = xv[i];
      float t  = __expf(-fabsf(x));                       // shared transcendental
      float rc = __builtin_amdgcn_rcpf(1.0f + t);
      float sg = (x >= 0.0f) ? rc : (1.0f - rc);          // sigmoid(x)
      float ng = fmaxf(x, 0.0f) + __logf(1.0f + t);       // softplus(x)
      float bn = (x >= 0.5f) ? 1.0f : 0.0f;               // binarized pred
      neg_s += ng; sig_s += sg; bin_s += bn;
      Apm[i] = (_Float16)x;
      Asg[i] = (_Float16)sg;
      Abn[i] = (_Float16)bn;
    }

    // ---- B tiles (tm is {0,1}: exact in f16)
    v16h Bm0, Bm1;
    if (USE_TMH) {
      Bm0 = *reinterpret_cast<const v16h*>(tmhRow0 + k + boff);   // 32B, aligned
      Bm1 = *reinterpret_cast<const v16h*>(tmhRow1 + k + boff);
    } else {
      const float4* pb0 = reinterpret_cast<const float4*>(tmRow0 + k + boff);
      const float4* pb1 = reinterpret_cast<const float4*>(tmRow1 + k + boff);
      float4 c0 = pb0[0], c1 = pb0[1], c2 = pb0[2], c3 = pb0[3];
      float4 d0 = pb1[0], d1 = pb1[1], d2 = pb1[2], d3 = pb1[3];
      float bv0[16] = { c0.x,c0.y,c0.z,c0.w, c1.x,c1.y,c1.z,c1.w,
                        c2.x,c2.y,c2.z,c2.w, c3.x,c3.y,c3.z,c3.w };
      float bv1[16] = { d0.x,d0.y,d0.z,d0.w, d1.x,d1.y,d1.z,d1.w,
                        d2.x,d2.y,d2.z,d2.w, d3.x,d3.y,d3.z,d3.w };
#pragma unroll
      for (int i = 0; i < 16; ++i) { Bm0[i] = (_Float16)bv0[i]; Bm1[i] = (_Float16)bv1[i]; }
    }

    // ---- 6 WMMAs: {pm, sig, bin} x {inst 0-15, inst 16-31}
    acc_pm0 = __builtin_amdgcn_wmma_f32_16x16x32_f16(false, Apm, false, Bm0, (short)0, acc_pm0, false, false);
    acc_pm1 = __builtin_amdgcn_wmma_f32_16x16x32_f16(false, Apm, false, Bm1, (short)0, acc_pm1, false, false);
    acc_sg0 = __builtin_amdgcn_wmma_f32_16x16x32_f16(false, Asg, false, Bm0, (short)0, acc_sg0, false, false);
    acc_sg1 = __builtin_amdgcn_wmma_f32_16x16x32_f16(false, Asg, false, Bm1, (short)0, acc_sg1, false, false);
    acc_bn0 = __builtin_amdgcn_wmma_f32_16x16x32_f16(false, Abn, false, Bm0, (short)0, acc_bn0, false, false);
    acc_bn1 = __builtin_amdgcn_wmma_f32_16x16x32_f16(false, Abn, false, Bm1, (short)0, acc_bn1, false, false);
  }

  // ---- merge split-K partials. D layout: lane holds column n=r; VGPR j -> row m = half*8+j
  float* acc = ws + ACC_OFF;
#pragma unroll
  for (int j = 0; j < 8; ++j) {
    const int m   = half * 8 + j;
    const long i0 = ((long)(b * NQ + mt * 16 + m) * NINST + r) * 3;
    const long i1 = ((long)(b * NQ + mt * 16 + m) * NINST + 16 + r) * 3;
    atomicAdd(acc + i0 + 0, acc_pm0[j]);
    atomicAdd(acc + i0 + 1, acc_sg0[j]);
    atomicAdd(acc + i0 + 2, acc_bn0[j]);
    atomicAdd(acc + i1 + 0, acc_pm1[j]);
    atomicAdd(acc + i1 + 1, acc_sg1[j]);
    atomicAdd(acc + i1 + 2, acc_bn1[j]);
  }
  float* row = ws + ROW_OFF + (long)(b * NQ + q) * 3;
  atomicAdd(row + 0, neg_s);
  atomicAdd(row + 1, sig_s);
  atomicAdd(row + 2, bin_s);
}

__global__ void hm_final(const float* __restrict__ pl, const int* __restrict__ tl,
                         const float* __restrict__ cw, const float* __restrict__ ws,
                         float* __restrict__ out) {
  const int t = blockIdx.x * blockDim.x + threadIdx.x;
  if (t >= B * NQ) return;
  const int b = t / NQ;

  // softmax over 19 classes
  const float* L = pl + (long)t * NCLS;
  float mx = L[0];
#pragma unroll
  for (int j = 1; j < NCLS; ++j) mx = fmaxf(mx, L[j]);
  float e[NCLS]; float s = 0.0f;
#pragma unroll
  for (int j = 0; j < NCLS; ++j) { e[j] = __expf(L[j] - mx); s += e[j]; }
  const float inv = 1.0f / s;

  const float* acc = ws + ACC_OFF + (long)t * NINST * 3;
  const float* row = ws + ROW_OFF + (long)t * 3;
  const float* tmr = ws + TMR_OFF + b * NINST;
  const float neg_s = row[0], sig_s = row[1], bin_s = row[2];
  const float cw0 = cw[0], cw1 = cw[1], cw2 = cw[2];

  float* Cout = out;
  float* Iout = out + (long)B * NQ * NINST;
  float bestC = 3.4e38f; int bestM = 0; float bestIou = 0.0f;

  for (int m = 0; m < NINST; ++m) {
    const float pm_dot = acc[m * 3 + 0];
    const float sg_dot = acc[m * 3 + 1];
    const float bn_dot = acc[m * 3 + 2];
    const float tms    = tmr[m];

    const float cclass = -e[tl[b * NINST + m]] * inv;
    const float cmask  = (neg_s - pm_dot) * (1.0f / (float)NPTS);
    const float cdice  = 1.0f - (2.0f * sg_dot + 1.0f) / (sig_s + tms + 1.0f);
    const float Cv     = cw0 * cclass + cw1 * cmask + cw2 * cdice;

    const float inter = bn_dot;
    const float iou   = inter / (bin_s + tms - inter + 1e-6f);

    Cout[(long)t * NINST + m] = Cv;
    Iout[(long)t * NINST + m] = iou;
    if (Cv < bestC) { bestC = Cv; bestM = m; bestIou = iou; }  // argmin, first occurrence
  }

  const long base = 2L * B * NQ * NINST;
  out[base + t] = (float)bestM;                                        // matches
  out[base + B * NQ + t] = (bestIou >= 0.6f) ? 1.0f
                         : ((bestIou >= 0.4f) ? -1.0f : 0.0f);         // match_labels
}

extern "C" void kernel_launch(void* const* d_in, const int* in_sizes, int n_in,
                              void* d_out, int out_size, void* d_ws, size_t ws_size,
                              hipStream_t stream) {
  const float* pl = (const float*)d_in[0];   // (B, NQ, NCLS)
  const float* pm = (const float*)d_in[1];   // (B, NQ, NPTS)
  const int*   tl = (const int*)  d_in[2];   // (B, NINST)
  const float* tm = (const float*)d_in[3];   // (B, NINST, NPTS)
  const float* cw = (const float*)d_in[4];   // (3,)
  float* ws  = (float*)d_ws;
  float* out = (float*)d_out;

  // f16 tm copy only if workspace is large enough (decision is fixed per harness config)
  const bool useTmh = ws_size >= (size_t)WS_REQ_TMH;
  _Float16* tmh = useTmh ? reinterpret_cast<_Float16*>(ws + TMH_OFF_F) : nullptr;

  hm_ws_zero<<<(WS_FLOATS + 255) / 256, 256, 0, stream>>>(ws);
  hm_tm_prep<<<B * NINST, 256, 0, stream>>>(tm, ws, tmh);
  if (useTmh) {
    hm_wmma_kernel<true><<<B * MTILES * KSPLIT, 32, 0, stream>>>(pm, tm, tmh, ws);
  } else {
    hm_wmma_kernel<false><<<B * MTILES * KSPLIT, 32, 0, stream>>>(pm, tm, tmh, ws);
  }
  hm_final<<<(B * NQ + 127) / 128, 128, 0, stream>>>(pl, tl, cw, ws, out);
}
